// MambaLayer_11373073400057
// MI455X (gfx1250) — compile-verified, matrix operand order checked by reading, not run on hardware
//
#include <hip/hip_runtime.h>
#include <hip/hip_bf16.h>

// ---------------------------------------------------------------------------
// Mamba layer for MI455X (gfx1250, wave32, WMMA + Tensor Data Mover).
// ---------------------------------------------------------------------------

#define D_MODEL 512
#define D_STATE 128
#define D_CONV  4
#define D_INNER 1024
#define DT_RANK 32
#define BATCH   4
#define SEQ     2048
#define NROWS   (BATCH * SEQ)          // 8192

typedef __attribute__((ext_vector_type(8)))  float        v8f;
typedef __attribute__((ext_vector_type(8)))  __bf16       v8bf;
typedef __attribute__((ext_vector_type(16))) __bf16       v16bf;
typedef __attribute__((ext_vector_type(4)))  unsigned int v4u;
typedef __attribute__((ext_vector_type(4)))  int          v4i;
typedef __attribute__((ext_vector_type(8)))  int          v8i;

#if __has_builtin(__builtin_amdgcn_tensor_load_to_lds) && \
    __has_builtin(__builtin_amdgcn_s_wait_tensorcnt)
#define HAVE_TDM 1
#else
#define HAVE_TDM 0
#warning "gfx1250 TDM builtins unavailable: falling back to cooperative LDS copy"
#endif

// ---------------------------------------------------------------------------
// fp32 -> bf16 cast
// ---------------------------------------------------------------------------
__global__ void cast_f32_to_bf16(const float* __restrict__ in,
                                 __bf16* __restrict__ out, int n) {
    int i = blockIdx.x * blockDim.x + threadIdx.x;
    if (i < n) out[i] = (__bf16)in[i];
}

// ---------------------------------------------------------------------------
// LayerNorm over D_MODEL=512, one block (256 thr) per row; bf16 output.
// ---------------------------------------------------------------------------
__global__ void layernorm_bf16(const float* __restrict__ x,
                               const float* __restrict__ g,
                               const float* __restrict__ b,
                               __bf16* __restrict__ out) {
    __shared__ float s[256];
    int row = blockIdx.x;
    int tid = threadIdx.x;
    const float* xr = x + (size_t)row * D_MODEL;
    float v0 = xr[tid];
    float v1 = xr[tid + 256];

    s[tid] = v0 + v1;
    __syncthreads();
    for (int st = 128; st > 0; st >>= 1) {
        if (tid < st) s[tid] += s[tid + st];
        __syncthreads();
    }
    float mu = s[0] * (1.0f / D_MODEL);
    __syncthreads();

    float d0 = v0 - mu, d1 = v1 - mu;
    s[tid] = d0 * d0 + d1 * d1;
    __syncthreads();
    for (int st = 128; st > 0; st >>= 1) {
        if (tid < st) s[tid] += s[tid + st];
        __syncthreads();
    }
    float inv = rsqrtf(s[0] * (1.0f / D_MODEL) + 1e-5f);

    __bf16* orow = out + (size_t)row * D_MODEL;
    orow[tid]       = (__bf16)(d0 * inv * g[tid]       + b[tid]);
    orow[tid + 256] = (__bf16)(d1 * inv * g[tid + 256] + b[tid + 256]);
}

// ---------------------------------------------------------------------------
// Fragment load matching the CDNA5 16-bit A/B WMMA VGPR layout:
//   lane<16  : row (lane&15), K = {k..k+7,  k+16..k+23}
//   lane>=16 : row (lane&15), K = {k+8..k+15, k+24..k+31}
// ---------------------------------------------------------------------------
__device__ inline v16bf load_k32(const __bf16* __restrict__ rowp, int k, int half) {
    v8bf lo = *(const v8bf*)(rowp + k + half * 8);
    v8bf hi = *(const v8bf*)(rowp + k + 16 + half * 8);
    v16bf v;
#pragma unroll
    for (int i = 0; i < 8; ++i) { v[i] = lo[i]; v[i + 8] = hi[i]; }
    return v;
}

__device__ inline float apply_epi(float xv, int EPI) {
    if (EPI == 2) return 0.5f * xv * (1.0f + erff(xv * 0.70710678118654752f));
    if (EPI == 3) return (xv > 20.0f) ? xv : log1pf(__expf(xv));
    return xv;
}

// ---------------------------------------------------------------------------
// Simple WMMA GEMM (one 16x16 tile/wave) - used for N=288 (x_proj).
// out[M,N] = epi(A[M,K] @ W[N,K]^T + bias)
// ---------------------------------------------------------------------------
template <int EPI>
__global__ void wmma_gemm(const __bf16* __restrict__ A, int lda,
                          const __bf16* __restrict__ W, int ldw,
                          const float* __restrict__ bias,
                          float* __restrict__ outF,
                          __bf16* __restrict__ outB,
                          int Mt, int Nt, int K, int ldo) {
    int wave = threadIdx.x >> 5;
    int lane = threadIdx.x & 31;
    long t = (long)blockIdx.x * (blockDim.x >> 5) + wave;
    if (t >= (long)Mt * Nt) return;           // wave-uniform
    int mt = (int)(t / Nt);
    int nt = (int)(t % Nt);

    int half = lane >> 4;
    int r    = lane & 15;
    const __bf16* aRow = A + (size_t)(mt * 16 + r) * lda;
    const __bf16* wRow = W + (size_t)(nt * 16 + r) * ldw;

    v8f c = {0.f, 0.f, 0.f, 0.f, 0.f, 0.f, 0.f, 0.f};
    for (int k = 0; k < K; k += 32) {
        if (k + 64 <= K) {
            __builtin_prefetch(aRow + k + 32, 0, 1);
            __builtin_prefetch(wRow + k + 32, 0, 1);
        }
        v16bf av = load_k32(aRow, k, half);
        v16bf bv = load_k32(wRow, k, half);
        c = __builtin_amdgcn_wmma_f32_16x16x32_bf16(
                false, av, false, bv, (short)0, c, false, false);
    }

    int col = nt * 16 + r;
    int m0  = mt * 16;
    float bv = bias ? bias[col] : 0.0f;
#pragma unroll
    for (int v = 0; v < 8; ++v) {
        int row = m0 + v + half * 8;
        float xv = apply_epi(c[v] + bv, EPI);
        if (outF) outF[(size_t)row * ldo + col] = xv;
        if (outB) outB[(size_t)row * ldo + col] = (__bf16)xv;
    }
}

// ---------------------------------------------------------------------------
// Blocked WMMA GEMM with double-buffered TDM-staged A tiles.
//   block tile : 32 rows x 512 cols (8 waves; wave tile 32x64 = 2x4 frags)
//   A tile (32 x CHUNK bf16) is DMA'd into LDS by the Tensor Data Mover,
//   +4-dword row padding (bank-conflict-free, 16B-aligned rows), while the
//   previous chunk is being consumed (s_wait_tensorcnt 1 pipeline).
// Requires: M % 32 == 0, N % 512 == 0, K % 32 == 0, (lda/2) integral.
// ---------------------------------------------------------------------------
#define CHUNKK 128                         // bf16 elements per staged K-chunk
#define PAD_EL 8                           // 4 dwords of LDS padding per row
#define TILE_EL (32 * (CHUNKK + PAD_EL))

#if HAVE_TDM
// Issue a TDM 2D tile load: 32 rows x (chunk/2) dwords from gsrc into LDS.
__device__ inline void tdm_issue_tile(const __bf16* gsrc, unsigned ldsAddr,
                                      int chunk, int K, int lda) {
    unsigned long long ga = (unsigned long long)gsrc;
    unsigned tile0 = (unsigned)(chunk >> 1);              // dwords per row
    unsigned td0   = (unsigned)(K >> 1);                  // dwords
    unsigned long long s0 = (unsigned long long)(lda >> 1); // dwords
    // pad_interval code: interval = chunk/2 dwords = 2^(code+1)
    unsigned padcode = (unsigned)(31 - __builtin_clz((unsigned)(chunk >> 1))) - 1u;

    v4u g0;
    g0[0] = 1u;                                           // count=1
    g0[1] = ldsAddr;
    g0[2] = (unsigned)ga;
    g0[3] = ((unsigned)(ga >> 32) & 0x01FFFFFFu) | (2u << 30);  // type=2
    v8i g1;
    g1[0] = (int)((2u << 16)            // data_size = 4B
                | (1u << 20)            // pad_enable
                | (padcode << 22)       // pad_interval
                | (3u << 25));          // pad_amount = 4 dwords
    g1[1] = (int)((td0 & 0xFFFFu) << 16);
    g1[2] = (int)((td0 >> 16) | (32u << 16));             // tensor_dim1 = 32
    g1[3] = (int)(tile0 << 16);                           // tile_dim0
    g1[4] = 32;                                           // tile_dim1 = 32
    g1[5] = (int)(unsigned)s0;
    g1[6] = (int)((unsigned)(s0 >> 32) & 0xFFFFu);
    g1[7] = 0;
    v4i z4 = {0, 0, 0, 0};
#if defined(__clang_major__) && (__clang_major__ >= 23)
    v8i z8 = {0, 0, 0, 0, 0, 0, 0, 0};
    __builtin_amdgcn_tensor_load_to_lds(g0, g1, z4, z4, z8, 0);
#else
    __builtin_amdgcn_tensor_load_to_lds(g0, g1, z4, z4, 0);
#endif
}
#endif

template <int EPI>
__global__ __launch_bounds__(256) void wmma_gemm_blocked(
        const __bf16* __restrict__ A, int lda,
        const __bf16* __restrict__ W, int ldw,
        const float* __restrict__ bias,
        float* __restrict__ outF, __bf16* __restrict__ outB,
        int Nb, int K, int ldo) {          // Nb = N/512
    __shared__ __bf16 smem[2][TILE_EL];

    int wave = threadIdx.x >> 5;
    int lane = threadIdx.x & 31;
    int mb   = blockIdx.x / Nb;            // 32-row block
    int nb   = blockIdx.x % Nb;            // 512-col block
    int half = lane >> 4;
    int r    = lane & 15;

    const int chunk   = (K < CHUNKK) ? K : CHUNKK;
    const int stride  = chunk + PAD_EL;    // LDS row stride (elements)
    const int nchunks = K / chunk;
    int n0 = nb * 512 + wave * 64;
    const __bf16* aBase = A + (size_t)mb * 32 * lda;

    v8f acc[8];
#pragma unroll
    for (int i = 0; i < 8; ++i) acc[i] = (v8f){0.f,0.f,0.f,0.f,0.f,0.f,0.f,0.f};

#if HAVE_TDM
    if (wave == 0) {                       // prime the pipeline
        tdm_issue_tile(aBase, (unsigned)(size_t)(void*)&smem[0][0],
                       chunk, K, lda);
    }
#endif

    for (int ci = 0; ci < nchunks; ++ci) {
        int k0 = ci * chunk;
#if HAVE_TDM
        if (wave == 0) {
            if (ci + 1 < nchunks) {        // overlap DMA of next chunk
                tdm_issue_tile(aBase + (size_t)(k0 + chunk),
                               (unsigned)(size_t)(void*)&smem[(ci + 1) & 1][0],
                               chunk, K, lda);
                __builtin_amdgcn_s_wait_tensorcnt((short)1);  // chunk ci done
            } else {
                __builtin_amdgcn_s_wait_tensorcnt((short)0);
            }
        }
#else
        {   // cooperative copy fallback, same padded layout
            int vecs = (32 * chunk) >> 3;                   // 8-elem units
            int perrow = chunk >> 3;
            for (int u = threadIdx.x; u < vecs; u += 256) {
                int rr = u / perrow, cc = u % perrow;
                *(v8bf*)(&smem[ci & 1][0] + rr * stride + cc * 8) =
                    *(const v8bf*)(aBase + (size_t)rr * lda + k0 + cc * 8);
            }
        }
#endif
        __syncthreads();

        const __bf16* sbuf = &smem[ci & 1][0];
        for (int kk = 0; kk < chunk; kk += 32) {
            v16bf a0 = load_k32(sbuf + r * stride,        kk, half);
            v16bf a1 = load_k32(sbuf + (16 + r) * stride, kk, half);
#pragma unroll
            for (int j = 0; j < 4; ++j) {
                const __bf16* wRow = W + (size_t)(n0 + j * 16 + r) * ldw;
                v16bf bv = load_k32(wRow, k0 + kk, half);
                acc[j]     = __builtin_amdgcn_wmma_f32_16x16x32_bf16(
                                 false, a0, false, bv, (short)0, acc[j], false, false);
                acc[4 + j] = __builtin_amdgcn_wmma_f32_16x16x32_bf16(
                                 false, a1, false, bv, (short)0, acc[4 + j], false, false);
            }
        }
        __syncthreads();   // buffer (ci&1) free before TDM reuses it (ci+2)
    }

    // ---- epilogue ----
#pragma unroll
    for (int mi = 0; mi < 2; ++mi) {
#pragma unroll
        for (int j = 0; j < 4; ++j) {
            v8f c = acc[mi * 4 + j];
            int col = n0 + j * 16 + r;
            float bv = bias ? bias[col] : 0.0f;
#pragma unroll
            for (int v = 0; v < 8; ++v) {
                int row = mb * 32 + mi * 16 + v + half * 8;
                float xv = apply_epi(c[v] + bv, EPI);
                if (outF) outF[(size_t)row * ldo + col] = xv;
                if (outB) outB[(size_t)row * ldo + col] = (__bf16)xv;
            }
        }
    }
}

// ---------------------------------------------------------------------------
// Causal depthwise conv (width 4) + bias + SiLU on xm = xz[:, :D_INNER].
// ---------------------------------------------------------------------------
__global__ void dwconv_silu(const float* __restrict__ xz,
                            const float* __restrict__ w,
                            const float* __restrict__ cb,
                            float* __restrict__ xcF,
                            __bf16* __restrict__ xcB) {
    size_t i = (size_t)blockIdx.x * blockDim.x + threadIdx.x;
    if (i >= (size_t)NROWS * D_INNER) return;
    int d      = (int)(i & (D_INNER - 1));
    size_t row = i >> 10;                     // b*SEQ + l
    int l      = (int)(row & (SEQ - 1));

    float acc = cb[d];
    const float* wd = w + d * D_CONV;
#pragma unroll
    for (int j = 0; j < D_CONV; ++j) {
        int ls = l - (D_CONV - 1) + j;
        if (ls >= 0)
            acc += wd[j] * xz[(row + (size_t)(ls - l)) * (2 * D_INNER) + d];
    }
    float s = acc * (1.0f / (1.0f + __expf(-acc)));   // SiLU
    xcF[i] = s;
    xcB[i] = (__bf16)s;
}

// ---------------------------------------------------------------------------
// Selective scan. One wave covers 2 channels (16 lanes each); each lane owns
// 8 states (n = (lane&15)*8 .. +7) -> 8-deep ILP hides v_exp latency, only a
// 4-step shuffle reduction per output.
// ---------------------------------------------------------------------------
__global__ void selective_scan(const float* __restrict__ u,      // (rows, Din)
                               const float* __restrict__ dt,     // (rows, Din)
                               const float* __restrict__ xdb,    // (rows, 288)
                               const float* __restrict__ A_log,  // (Din, N)
                               const float* __restrict__ Dp,     // (Din)
                               float* __restrict__ y) {          // (rows, Din)
    int wave = threadIdx.x >> 5;
    int lane = threadIdx.x & 31;
    int gw   = blockIdx.x * (blockDim.x >> 5) + wave;   // 0..2047
    int b    = gw >> 9;                                  // 512 pairs per batch
    int d    = ((gw & 511) << 1) + (lane >> 4);
    int ng   = lane & 15;                                // n-group

    const float* ap = A_log + (size_t)d * D_STATE + ng * 8;
    float4 alo = *(const float4*)(ap);
    float4 ahi = *(const float4*)(ap + 4);
    float A0 = -__expf(alo.x), A1 = -__expf(alo.y);
    float A2 = -__expf(alo.z), A3 = -__expf(alo.w);
    float A4 = -__expf(ahi.x), A5 = -__expf(ahi.y);
    float A6 = -__expf(ahi.z), A7 = -__expf(ahi.w);
    float h0=0.f,h1=0.f,h2=0.f,h3=0.f,h4=0.f,h5=0.f,h6=0.f,h7=0.f;
    float Dv = Dp[d];

    size_t rowbase = (size_t)b * SEQ;
    for (int l = 0; l < SEQ; ++l) {
        size_t row = rowbase + l;
        float uv  = u [row * D_INNER + d];
        float dtv = dt[row * D_INNER + d];
        const float* xr = xdb + row * (DT_RANK + 2 * D_STATE);
        float4 B0 = *(const float4*)(xr + DT_RANK + ng * 8);
        float4 B1 = *(const float4*)(xr + DT_RANK + ng * 8 + 4);
        float4 C0 = *(const float4*)(xr + DT_RANK + D_STATE + ng * 8);
        float4 C1 = *(const float4*)(xr + DT_RANK + D_STATE + ng * 8 + 4);

        float dtu = dtv * uv;
        h0 = h0 * __expf(dtv * A0) + dtu * B0.x;
        h1 = h1 * __expf(dtv * A1) + dtu * B0.y;
        h2 = h2 * __expf(dtv * A2) + dtu * B0.z;
        h3 = h3 * __expf(dtv * A3) + dtu * B0.w;
        h4 = h4 * __expf(dtv * A4) + dtu * B1.x;
        h5 = h5 * __expf(dtv * A5) + dtu * B1.y;
        h6 = h6 * __expf(dtv * A6) + dtu * B1.z;
        h7 = h7 * __expf(dtv * A7) + dtu * B1.w;

        float acc = h0*C0.x + h1*C0.y + h2*C0.z + h3*C0.w
                  + h4*C1.x + h5*C1.y + h6*C1.z + h7*C1.w;
#pragma unroll
        for (int off = 1; off < 16; off <<= 1)           // reduce 16 lanes
            acc += __shfl_xor(acc, off, 32);
        if (ng == 0) y[row * D_INNER + d] = acc + uv * Dv;
    }
}

// ---------------------------------------------------------------------------
// y * silu(z) -> bf16 (z = xz[:, D_INNER:2*D_INNER])
// ---------------------------------------------------------------------------
__global__ void gate_silu_bf16(const float* __restrict__ xz,
                               const float* __restrict__ y,
                               __bf16* __restrict__ out) {
    size_t i = (size_t)blockIdx.x * blockDim.x + threadIdx.x;
    if (i >= (size_t)NROWS * D_INNER) return;
    int d      = (int)(i & (D_INNER - 1));
    size_t row = i >> 10;
    float z = xz[row * (2 * D_INNER) + D_INNER + d];
    float s = z * (1.0f / (1.0f + __expf(-z)));
    out[i] = (__bf16)(y[i] * s);
}

// ---------------------------------------------------------------------------
// Host
// ---------------------------------------------------------------------------
static inline int cdiv(long a, long b) { return (int)((a + b - 1) / b); }

extern "C" void kernel_launch(void* const* d_in, const int* in_sizes, int n_in,
                              void* d_out, int out_size, void* d_ws, size_t ws_size,
                              hipStream_t stream) {
    (void)in_sizes; (void)n_in; (void)out_size; (void)ws_size;
    const float* x         = (const float*)d_in[0];
    const float* ln_g      = (const float*)d_in[1];
    const float* ln_b      = (const float*)d_in[2];
    const float* in_proj_w = (const float*)d_in[3];
    const float* conv_w    = (const float*)d_in[4];
    const float* conv_b    = (const float*)d_in[5];
    const float* x_proj_w  = (const float*)d_in[6];
    const float* dt_proj_w = (const float*)d_in[7];
    const float* dt_proj_b = (const float*)d_in[8];
    const float* A_log     = (const float*)d_in[9];
    const float* D_param   = (const float*)d_in[10];
    const float* out_proj_w= (const float*)d_in[11];
    const float* mlp_w1    = (const float*)d_in[12];
    const float* mlp_b1    = (const float*)d_in[13];
    const float* mlp_w2    = (const float*)d_in[14];
    const float* mlp_b2    = (const float*)d_in[15];

    // ---- workspace layout ----
    char* ws = (char*)d_ws;
    size_t off = 0;
    auto alloc = [&](size_t bytes) -> void* {
        void* p = ws + off;
        off += (bytes + 255) & ~(size_t)255;
        return p;
    };
    __bf16* wInB  = (__bf16*)alloc((size_t)2 * D_INNER * D_MODEL * 2);
    __bf16* wXpB  = (__bf16*)alloc((size_t)(DT_RANK + 2 * D_STATE) * D_INNER * 2);
    __bf16* wDtB  = (__bf16*)alloc((size_t)D_INNER * DT_RANK * 2);
    __bf16* wOutB = (__bf16*)alloc((size_t)D_MODEL * D_INNER * 2);
    __bf16* wM1B  = (__bf16*)alloc((size_t)4 * D_MODEL * D_MODEL * 2);
    __bf16* wM2B  = (__bf16*)alloc((size_t)D_MODEL * 4 * D_MODEL * 2);
    __bf16* hB    = (__bf16*)alloc((size_t)NROWS * D_MODEL * 2);
    float*  xzF   = (float*) alloc((size_t)NROWS * 2 * D_INNER * 4);
    float*  xcF   = (float*) alloc((size_t)NROWS * D_INNER * 4);
    __bf16* xcB   = (__bf16*)alloc((size_t)NROWS * D_INNER * 2);
    float*  xdbF  = (float*) alloc((size_t)NROWS * (DT_RANK + 2 * D_STATE) * 4);
    __bf16* xdbB  = (__bf16*)alloc((size_t)NROWS * (DT_RANK + 2 * D_STATE) * 2);
    float*  dtF   = (float*) alloc((size_t)NROWS * D_INNER * 4);
    float*  yF    = (float*) alloc((size_t)NROWS * D_INNER * 4);
    __bf16* yB    = (__bf16*)alloc((size_t)NROWS * D_INNER * 2);
    __bf16* outBf = (__bf16*)alloc((size_t)NROWS * D_MODEL * 2);
    __bf16* hmB   = (__bf16*)alloc((size_t)NROWS * 4 * D_MODEL * 2);

    // ---- weight casts ----
    auto cast = [&](const float* src, __bf16* dst, long n) {
        cast_f32_to_bf16<<<cdiv(n, 256), 256, 0, stream>>>(src, dst, (int)n);
    };
    cast(in_proj_w,  wInB,  (long)2 * D_INNER * D_MODEL);
    cast(x_proj_w,   wXpB,  (long)(DT_RANK + 2 * D_STATE) * D_INNER);
    cast(dt_proj_w,  wDtB,  (long)D_INNER * DT_RANK);
    cast(out_proj_w, wOutB, (long)D_MODEL * D_INNER);
    cast(mlp_w1,     wM1B,  (long)4 * D_MODEL * D_MODEL);
    cast(mlp_w2,     wM2B,  (long)D_MODEL * 4 * D_MODEL);

    // ---- 1. LayerNorm -> bf16 ----
    layernorm_bf16<<<NROWS, 256, 0, stream>>>(x, ln_g, ln_b, hB);

    const int Mb = NROWS / 32;   // 256 row-blocks

    // ---- 2. in_proj: xz = h @ Win^T  (K=512, N=2048) ----
    wmma_gemm_blocked<0><<<Mb * 4, 256, 0, stream>>>(
        hB, D_MODEL, wInB, D_MODEL, nullptr, xzF, nullptr,
        4, D_MODEL, 2 * D_INNER);

    // ---- 3. depthwise conv + SiLU ----
    dwconv_silu<<<cdiv((long)NROWS * D_INNER, 256), 256, 0, stream>>>(
        xzF, conv_w, conv_b, xcF, xcB);

    // ---- 4. x_proj: xdb = xc @ Wxp^T  (K=1024, N=288) -> simple kernel ----
    wmma_gemm<0><<<cdiv((long)(NROWS / 16) * 18, 8), 256, 0, stream>>>(
        xcB, D_INNER, wXpB, D_INNER, nullptr, xdbF, xdbB,
        NROWS / 16, 18, D_INNER, DT_RANK + 2 * D_STATE);

    // ---- 5. dt = softplus(dt_lo @ Wdt^T + b)  (K=32, N=1024) ----
    wmma_gemm_blocked<3><<<Mb * 2, 256, 0, stream>>>(
        xdbB, DT_RANK + 2 * D_STATE, wDtB, DT_RANK, dt_proj_b, dtF, nullptr,
        2, DT_RANK, D_INNER);

    // ---- 6. selective scan ----
    selective_scan<<<(BATCH * D_INNER / 2) / 8, 256, 0, stream>>>(
        xcF, dtF, xdbF, A_log, D_param, yF);

    // ---- 7. gating: y * silu(z) -> bf16 ----
    gate_silu_bf16<<<cdiv((long)NROWS * D_INNER, 256), 256, 0, stream>>>(
        xzF, yF, yB);

    // ---- 8. out_proj (K=1024, N=512) -> bf16 ----
    wmma_gemm_blocked<0><<<Mb * 1, 256, 0, stream>>>(
        yB, D_INNER, wOutB, D_INNER, nullptr, nullptr, outBf,
        1, D_INNER, D_MODEL);

    // ---- 9. MLP fc1 + exact GELU (K=512, N=2048) -> bf16 ----
    wmma_gemm_blocked<2><<<Mb * 4, 256, 0, stream>>>(
        outBf, D_MODEL, wM1B, D_MODEL, mlp_b1, nullptr, hmB,
        4, D_MODEL, 4 * D_MODEL);

    // ---- 10. MLP fc2 + bias (K=2048, N=512) -> d_out fp32 ----
    wmma_gemm_blocked<0><<<Mb * 1, 256, 0, stream>>>(
        hmB, 4 * D_MODEL, wM2B, 4 * D_MODEL, mlp_b2, (float*)d_out, nullptr,
        1, 4 * D_MODEL, D_MODEL);
}
